// Attn_6227702579874
// MI455X (gfx1250) — matrix operand order repaired and checked by reading, not compiled, not run on hardware
//
#include <hip/hip_runtime.h>
#include <hip/hip_bf16.h>
#include <math.h>

// ---------------------------------------------------------------------------
// Types for CDNA5 WMMA (wave32, 16x16x32 bf16 -> f32)
// ---------------------------------------------------------------------------
typedef __bf16 bf16;
typedef __attribute__((ext_vector_type(16))) __bf16 v16bf;
typedef __attribute__((ext_vector_type(8)))  __bf16 v8bf;
typedef __attribute__((ext_vector_type(4)))  __bf16 v4bf;
typedef __attribute__((ext_vector_type(8)))  float  v8f;
typedef __attribute__((ext_vector_type(4)))  float  v4f;
// 2-byte-aligned vector view for +/-1 shifted reads of padded bf16 maps
typedef v8bf v8bf_u __attribute__((aligned(2)));

#define LDK 56  // padded LDS leading dim (elems): 112B rows (16B aligned, conflict-friendly)

__device__ __forceinline__ v8f vzero8() {
    v8f z;
#pragma unroll
    for (int i = 0; i < 8; ++i) z[i] = 0.0f;
    return z;
}

// A fragment (16x32): M=lane&15; lanes<16 -> K 0..7,16..23; lanes>=16 -> 8..15,24..31
__device__ __forceinline__ v16bf frag_a(const bf16* t, int lane) {
    const bf16* p = t + (lane & 15) * LDK + ((lane >> 4) << 3);
    v16bf f;
    *(v8bf*)&f       = *(const v8bf*)p;
    *((v8bf*)&f + 1) = *(const v8bf*)(p + 16);
    return f;
}

// B fragment (32x16 stored N-major): N=lane&15; lanes<16 -> K 0..7,16..23; lanes>=16 -> 8..15,24..31
__device__ __forceinline__ v16bf frag_b(const bf16* t, int lane) {
    const bf16* p = t + (lane & 15) * LDK + ((lane >> 4) << 3);
    v16bf f;
    *(v8bf*)&f       = *(const v8bf*)p;
    *((v8bf*)&f + 1) = *(const v8bf*)(p + 16);
    return f;
}

__device__ __forceinline__ v8f wmma_bf(v16bf a, v16bf b, v8f c) {
    return __builtin_amdgcn_wmma_f32_16x16x32_bf16(false, a, false, b, (short)0, c, false, false);
}

// ---------------------------------------------------------------------------
// conv1x1 as GEMM: Ypad(128, (QW+2)^2) = W(128,256) * X(256,QW^2) + bias
// All geometry compile-time (QSH in {5,6}); output padded (zero borders).
// ---------------------------------------------------------------------------
template <int QSH>
__global__ __launch_bounds__(256) void k_conv1x1(
    const float* __restrict__ W, const float* __restrict__ bias,
    const bf16* __restrict__ X, bf16* __restrict__ Y)
{
    constexpr int QW = 1 << QSH, PW = QW + 2, PN = PW * PW;
    constexpr int N = QW * QW, C = 256;
    __shared__ __align__(16) bf16 As[128 * LDK];
    __shared__ __align__(16) bf16 Bs[128 * LDK];
    const int b = blockIdx.z;
    const int pbase = blockIdx.x * 128;
    const bf16* Xb = X + (size_t)b * C * N;
    bf16* Yb = Y + (size_t)b * 128 * PN;
    const int tx = threadIdx.x;
    const int w = tx >> 5, lane = tx & 31;
    const int wm = w >> 1, wn = w & 1;  // 4x2 waves, each 32(M) x 64(N)
    v8f acc[2][4];
    for (int i = 0; i < 2; ++i) for (int j = 0; j < 4; ++j) acc[i][j] = vzero8();

    for (int kc = 0; kc < C; kc += 32) {
        __syncthreads();
#pragma unroll
        for (int e = 0; e < 4; ++e) {  // A: 128(m) x 32(k) fp32->bf16, vector loads
            int vi = e * 256 + tx;
            int m = vi >> 3, k4 = (vi & 7) * 4;
            v4f wv = *(const v4f*)(W + (size_t)m * C + kc + k4);
            v4bf t;
            t[0] = (bf16)wv[0]; t[1] = (bf16)wv[1]; t[2] = (bf16)wv[2]; t[3] = (bf16)wv[3];
            *(v4bf*)&As[m * LDK + k4] = t;
        }
#pragma unroll
        for (int e = 0; e < 2; ++e) {  // B: transpose to n-major, v8 global loads
            int vi = e * 256 + tx;
            int k = vi >> 4, n8 = (vi & 15) * 8;
            v8bf v = *(const v8bf*)(Xb + (size_t)(kc + k) * N + pbase + n8);
#pragma unroll
            for (int j = 0; j < 8; ++j) Bs[(n8 + j) * LDK + k] = v[j];
        }
        __syncthreads();
        v16bf af[2], bfg[4];
#pragma unroll
        for (int mt = 0; mt < 2; ++mt) af[mt] = frag_a(&As[(wm * 32 + mt * 16) * LDK], lane);
#pragma unroll
        for (int nt = 0; nt < 4; ++nt) bfg[nt] = frag_b(&Bs[(wn * 64 + nt * 16) * LDK], lane);
#pragma unroll
        for (int mt = 0; mt < 2; ++mt)
#pragma unroll
            for (int nt = 0; nt < 4; ++nt)
                acc[mt][nt] = wmma_bf(af[mt], bfg[nt], acc[mt][nt]);
    }
#pragma unroll
    for (int mt = 0; mt < 2; ++mt)
#pragma unroll
        for (int nt = 0; nt < 4; ++nt)
#pragma unroll
            for (int r = 0; r < 8; ++r) {
                int m = wm * 32 + mt * 16 + r + ((lane >> 4) << 3);
                int n = pbase + wn * 64 + nt * 16 + (lane & 15);
                int py = n >> QSH, px = n & (QW - 1);
                Yb[(size_t)m * PN + (py + 1) * PW + (px + 1)] =
                    (bf16)(acc[mt][nt][r] + bias[m]);
            }
}

// ---------------------------------------------------------------------------
// Patch correlation: S[q,p] = rscale[q] * sum_u sum_c Wm[c,q+u]*F[c,p+u]
// Wm padded (QW+2)^2 per channel; F padded 66x66. No predication.
// ---------------------------------------------------------------------------
template <int QSH>
__global__ __launch_bounds__(256) void k_corr(
    const bf16* __restrict__ Wm, const bf16* __restrict__ F,
    const float* __restrict__ rsc, float* __restrict__ S)
{
    constexpr int QW = 1 << QSH, PWq = QW + 2, PNq = PWq * PWq;
    __shared__ __align__(16) bf16 As[128 * LDK];
    __shared__ __align__(16) bf16 Bs[128 * LDK];
    const int qbase = blockIdx.y * 128, pbase = blockIdx.x * 128;
    const int tx = threadIdx.x;
    const int w = tx >> 5, lane = tx & 31, wm = w >> 1, wn = w & 1;
    v8f acc[2][4];
    for (int i = 0; i < 2; ++i) for (int j = 0; j < 4; ++j) acc[i][j] = vzero8();

    for (int u = 0; u < 9; ++u) {
        const int du = u / 3 - 1, dv = u % 3 - 1;
        for (int cb = 0; cb < 128; cb += 32) {
            __syncthreads();
#pragma unroll
            for (int e = 0; e < 2; ++e) {  // A: [q][c] shifted (padded) read of Wm
                int vi = e * 256 + tx;
                int c = vi >> 4, q8 = (vi & 15) * 8;
                int q = qbase + q8;
                int qy = q >> QSH, qx = q & (QW - 1);
                const bf16* g = Wm + (size_t)(cb + c) * PNq + (qy + 1 + du) * PWq + (qx + 1 + dv);
                v8bf v = *(const v8bf_u*)g;
#pragma unroll
                for (int j = 0; j < 8; ++j) As[(q8 + j) * LDK + c] = v[j];
            }
#pragma unroll
            for (int e = 0; e < 2; ++e) {  // B: [p][c] shifted (padded) read of F
                int vi = e * 256 + tx;
                int c = vi >> 4, p8 = (vi & 15) * 8;
                int p = pbase + p8;
                int py = p >> 6, px = p & 63;
                const bf16* g = F + (size_t)(cb + c) * 4356 + (py + 1 + du) * 66 + (px + 1 + dv);
                v8bf v = *(const v8bf_u*)g;
#pragma unroll
                for (int j = 0; j < 8; ++j) Bs[(p8 + j) * LDK + c] = v[j];
            }
            __syncthreads();
            v16bf af[2], bfg[4];
#pragma unroll
            for (int mt = 0; mt < 2; ++mt) af[mt] = frag_a(&As[(wm * 32 + mt * 16) * LDK], lane);
#pragma unroll
            for (int nt = 0; nt < 4; ++nt) bfg[nt] = frag_b(&Bs[(wn * 64 + nt * 16) * LDK], lane);
#pragma unroll
            for (int mt = 0; mt < 2; ++mt)
#pragma unroll
                for (int nt = 0; nt < 4; ++nt)
                    acc[mt][nt] = wmma_bf(af[mt], bfg[nt], acc[mt][nt]);
        }
    }
#pragma unroll
    for (int mt = 0; mt < 2; ++mt)
#pragma unroll
        for (int nt = 0; nt < 4; ++nt)
#pragma unroll
            for (int r = 0; r < 8; ++r) {
                int m = qbase + wm * 32 + mt * 16 + r + ((lane >> 4) << 3);
                int n = pbase + wn * 64 + nt * 16 + (lane & 15);
                S[(size_t)m * 4096 + n] = acc[mt][nt][r] * rsc[m];
            }
}

// ---------------------------------------------------------------------------
// Column softmax over q with mask logic, staged through LDS with CDNA5
// async global->LDS copies (ASYNCcnt). Writes padded probability matrix.
// ---------------------------------------------------------------------------
__global__ __launch_bounds__(256) void k_softmax(
    const float* __restrict__ S, const float* __restrict__ mm,
    bf16* __restrict__ Pm, int Q)
{
    __shared__ float smm[4096];
    __shared__ __align__(16) float tile[64 * 256];
    const int tx = threadIdx.x;
    const int p0 = blockIdx.x * 256;
    for (int i = tx; i < Q; i += 256) smm[i] = mm[i];

    auto stage = [&](int q0) {
        __syncthreads();  // protect previous tile consumers
#pragma unroll
        for (int e = 0; e < 16; ++e) {  // 64 rows x 256 cols f32 = 64KB
            int ci = e * 256 + tx;
            int row = ci >> 6, seg = ci & 63;
            unsigned lo = (unsigned)(size_t)(void*)&tile[row * 256 + seg * 4];
            const float* g = S + (size_t)(q0 + row) * 4096 + p0 + seg * 4;
            asm volatile("global_load_async_to_lds_b128 %0, %1, off"
                         :: "v"(lo), "v"(g) : "memory");
        }
        asm volatile("s_wait_asynccnt 0x0" ::: "memory");
        __syncthreads();
    };

    float mx = -1e30f;
    for (int q0 = 0; q0 < Q; q0 += 64) {
        stage(q0);
        for (int r = 0; r < 64; ++r)
            mx = fmaxf(mx, tile[r * 256 + tx] * smm[q0 + r] * 10.0f);
    }
    float sum = 0.0f;
    for (int q0 = 0; q0 < Q; q0 += 64) {
        stage(q0);
        for (int r = 0; r < 64; ++r)
            sum += expf(tile[r * 256 + tx] * smm[q0 + r] * 10.0f - mx);
    }
    const float inv = 1.0f / sum;
    const int p = p0 + tx;
    const int py = p >> 6, px = p & 63;
    const size_t poff = (size_t)(py + 1) * 66 + (px + 1);
    for (int q0 = 0; q0 < Q; q0 += 64) {
        stage(q0);
        for (int r = 0; r < 64; ++r) {
            float l = tile[r * 256 + tx] * smm[q0 + r] * 10.0f;
            float pr = expf(l - mx) * inv * smm[q0 + r];
            Pm[(size_t)(q0 + r) * 4356 + poff] = (bf16)fmaxf(pr, 1e-8f);
        }
    }
}

// ---------------------------------------------------------------------------
// Transposed conv: O[co,p] += 0.25 * sum_q V[co,q+u] * P[q,p-u], u=blockIdx.z
// V padded (QW+2)^2 per channel; P padded 66x66 per q-row. No predication.
// ---------------------------------------------------------------------------
template <int QSH>
__global__ __launch_bounds__(256) void k_convT(
    const bf16* __restrict__ V, const bf16* __restrict__ Pm,
    float* __restrict__ outp)
{
    constexpr int QW = 1 << QSH, PWq = QW + 2, PNq = PWq * PWq, QN = QW * QW;
    __shared__ __align__(16) bf16 As[128 * LDK];
    __shared__ __align__(16) bf16 Bs[128 * LDK];
    const int u = blockIdx.z, du = u / 3 - 1, dv = u % 3 - 1;
    const int pbase = blockIdx.x * 128;
    const int tx = threadIdx.x;
    const int w = tx >> 5, lane = tx & 31, wm = w >> 1, wn = w & 1;
    v8f acc[2][4];
    for (int i = 0; i < 2; ++i) for (int j = 0; j < 4; ++j) acc[i][j] = vzero8();

    for (int qb = 0; qb < QN; qb += 32) {
        __syncthreads();
#pragma unroll
        for (int e = 0; e < 2; ++e) {  // A: [co][k=q] — no transpose, b128 LDS stores
            int vi = e * 256 + tx;
            int co = vi >> 2, k8 = (vi & 3) * 8;
            int q = qb + k8;
            int qy = q >> QSH, qx = q & (QW - 1);
            const bf16* g = V + (size_t)co * PNq + (qy + 1 + du) * PWq + (qx + 1 + dv);
            v8bf v = *(const v8bf_u*)g;
            *(v8bf*)&As[co * LDK + k8] = v;
        }
#pragma unroll
        for (int e = 0; e < 2; ++e) {  // B: [p][k=q] = Pm[q, p-u], transpose
            int vi = e * 256 + tx;
            int k = vi >> 4, p8 = (vi & 15) * 8;
            int p = pbase + p8;
            int py = p >> 6, px = p & 63;
            const bf16* g = Pm + (size_t)(qb + k) * 4356 + (py + 1 - du) * 66 + (px + 1 - dv);
            v8bf v = *(const v8bf_u*)g;
#pragma unroll
            for (int j = 0; j < 8; ++j) Bs[(p8 + j) * LDK + k] = v[j];
        }
        __syncthreads();
        v16bf af[2], bfg[4];
#pragma unroll
        for (int mt = 0; mt < 2; ++mt) af[mt] = frag_a(&As[(wm * 32 + mt * 16) * LDK], lane);
#pragma unroll
        for (int nt = 0; nt < 4; ++nt) bfg[nt] = frag_b(&Bs[(wn * 64 + nt * 16) * LDK], lane);
#pragma unroll
        for (int mt = 0; mt < 2; ++mt)
#pragma unroll
            for (int nt = 0; nt < 4; ++nt)
                acc[mt][nt] = wmma_bf(af[mt], bfg[nt], acc[mt][nt]);
    }
#pragma unroll
    for (int mt = 0; mt < 2; ++mt)
#pragma unroll
        for (int nt = 0; nt < 4; ++nt)
#pragma unroll
            for (int r = 0; r < 8; ++r) {
                int m = wm * 32 + mt * 16 + r + ((lane >> 4) << 3);
                int n = pbase + wn * 64 + nt * 16 + (lane & 15);
                __hip_atomic_fetch_add(&outp[(size_t)m * 4096 + n],
                                       0.25f * acc[mt][nt][r],
                                       __ATOMIC_RELAXED, __HIP_MEMORY_SCOPE_AGENT);
            }
}

// ---------------------------------------------------------------------------
// Dilated 3x3 convs (rates 1,2,4,8 = blockIdx.y), leaky-relu, + x1 residual.
// attn map padded with 8-px zero border (max dilation shift).
// ---------------------------------------------------------------------------
__global__ __launch_bounds__(256) void k_dilconv(
    const float* __restrict__ gw, const float* __restrict__ gb,
    const bf16* __restrict__ attn, const float* __restrict__ x1,
    float* __restrict__ outp)
{
    __shared__ __align__(16) bf16 As[64 * LDK];
    __shared__ __align__(16) bf16 Bs[128 * LDK];
    const int b = blockIdx.z, ri = blockIdx.y, pbase = blockIdx.x * 128;
    const int rate = 1 << ri;  // 1,2,4,8
    const int tx = threadIdx.x, w = tx >> 5, lane = tx & 31;
    const int wm = w >> 2, wn = w & 3;  // 2x4 waves, each 32x32
    v8f acc[2][2];
    for (int i = 0; i < 2; ++i) for (int j = 0; j < 2; ++j) acc[i][j] = vzero8();
    const bf16* attnb = attn + (size_t)b * 256 * 6400;

    for (int u = 0; u < 9; ++u) {
        const int du = u / 3 - 1, dv = u % 3 - 1;
        for (int cb = 0; cb < 256; cb += 32) {
            __syncthreads();
#pragma unroll
            for (int e = 0; e < 8; ++e) {  // A: 64(o) x 32(c), stride-9 gather
                int idx = e * 256 + tx;
                int kk = idx & 31, m = idx >> 5;
                As[m * LDK + kk] = (bf16)gw[(size_t)((ri * 64 + m) * 256 + cb + kk) * 9 + u];
            }
#pragma unroll
            for (int e = 0; e < 2; ++e) {  // B: [p][c] dilated shifted (padded) read
                int vi = e * 256 + tx;
                int k = vi >> 4, n8 = (vi & 15) * 8;
                int p = pbase + n8;
                int py = p >> 6, px = p & 63;
                const bf16* g = attnb + (size_t)(cb + k) * 6400
                              + (py + 8 + rate * du) * 80 + (px + 8 + rate * dv);
                v8bf v = *(const v8bf_u*)g;
#pragma unroll
                for (int j = 0; j < 8; ++j) Bs[(n8 + j) * LDK + k] = v[j];
            }
            __syncthreads();
            v16bf af[2], bfg[2];
#pragma unroll
            for (int mt = 0; mt < 2; ++mt) af[mt] = frag_a(&As[(wm * 32 + mt * 16) * LDK], lane);
#pragma unroll
            for (int nt = 0; nt < 2; ++nt) bfg[nt] = frag_b(&Bs[(wn * 32 + nt * 16) * LDK], lane);
#pragma unroll
            for (int mt = 0; mt < 2; ++mt)
#pragma unroll
                for (int nt = 0; nt < 2; ++nt)
                    acc[mt][nt] = wmma_bf(af[mt], bfg[nt], acc[mt][nt]);
        }
    }
#pragma unroll
    for (int mt = 0; mt < 2; ++mt)
#pragma unroll
        for (int nt = 0; nt < 2; ++nt)
#pragma unroll
            for (int r = 0; r < 8; ++r) {
                int m = wm * 32 + mt * 16 + r + ((lane >> 4) << 3);  // 0..63
                int n = pbase + wn * 32 + nt * 16 + (lane & 15);
                int ch = ri * 64 + m;
                float v = acc[mt][nt][r] + gb[ri * 64 + m];
                v = v > 0.0f ? v : 0.2f * v;  // leaky relu
                size_t o = ((size_t)b * 256 + ch) * 4096 + n;
                outp[o] = v + x1[o];
            }
}

// ---------------------------------------------------------------------------
// Small helper kernels
// ---------------------------------------------------------------------------
__global__ void k_f32_to_bf16(const float* __restrict__ x, bf16* __restrict__ y, int n) {
    int i = blockIdx.x * 256 + threadIdx.x;
    if (i < n) y[i] = (bf16)x[i];
}

typedef __attribute__((ext_vector_type(4))) unsigned int v4u;
__global__ void k_zero128(v4u* __restrict__ p, int n16) {
    int i = blockIdx.x * 256 + threadIdx.x;
    if (i < n16) { v4u z; z[0]=0u; z[1]=0u; z[2]=0u; z[3]=0u; p[i] = z; }
}

// attn f32 (B,256,64,64) -> bf16 padded (B,256,80,80) interior
__global__ void k_attn_pack(const float* __restrict__ x, bf16* __restrict__ y) {
    int i = blockIdx.x * 256 + threadIdx.x;
    if (i >= 2 * 256 * 4096) return;
    int px = i & 63, py = (i >> 6) & 63, ch = (i >> 12) & 255, b = i >> 20;
    y[(size_t)(b * 256 + ch) * 6400 + (py + 8) * 80 + (px + 8)] = (bf16)x[i];
}

// bilinear 64x64 -> 32x32 (linspace endpoints), f32 -> bf16 (unpadded: conv input)
__global__ void k_resize32(const float* __restrict__ x2, bf16* __restrict__ y) {
    int i = blockIdx.x * 256 + threadIdx.x;
    if (i >= 2 * 256 * 1024) return;
    int ox = i & 31, oy = (i >> 5) & 31, c = (i >> 10) & 255, b = i >> 18;
    const float step = 63.0f / 31.0f;
    float ry = oy * step, rx = ox * step;
    int y0 = (int)floorf(ry); int y1 = y0 + 1 > 63 ? 63 : y0 + 1; float wy = ry - (float)y0;
    int x0 = (int)floorf(rx); int x1i = x0 + 1 > 63 ? 63 : x0 + 1; float wx = rx - (float)x0;
    const float* p = x2 + ((size_t)(b * 256 + c)) * 4096;
    float a  = p[y0 * 64 + x0] * (1.0f - wx) + p[y0 * 64 + x1i] * wx;
    float bb = p[y1 * 64 + x0] * (1.0f - wx) + p[y1 * 64 + x1i] * wx;
    y[i] = (bf16)(a * (1.0f - wy) + bb * wy);
}

// mm[b,q] = (sum of 3x3 zero-padded patch of (subsampled) mask)/9 > 0.5
__global__ void k_mm(const float* __restrict__ mask, float* __restrict__ mm, int qsh, int stride) {
    int QW = 1 << qsh, QN = QW * QW;
    int i = blockIdx.x * 256 + threadIdx.x;
    if (i >= 2 * QN) return;
    int b = i / QN, q = i % QN;
    int qy = q >> qsh, qx = q & (QW - 1);
    float s = 0.0f;
    for (int du = -1; du <= 1; ++du)
        for (int dv = -1; dv <= 1; ++dv) {
            int yy = qy + du, xx = qx + dv;
            if ((unsigned)yy < (unsigned)QW && (unsigned)xx < (unsigned)QW)
                s += mask[(size_t)b * 4096 + (yy * stride) * 64 + xx * stride];
        }
    mm[i] = (s / 9.0f) > 0.5f ? 1.0f : 0.0f;
}

// rscale[b,q] = 1/max(||patch_q(Wm)||, 1e-4); padded map -> branch-free
__global__ void k_rscale(const bf16* __restrict__ Wm, float* __restrict__ rsc, int qsh) {
    int QW = 1 << qsh, QN = QW * QW, PW = QW + 2, PN = PW * PW;
    int i = blockIdx.x * 256 + threadIdx.x;
    if (i >= 2 * QN) return;
    int b = i / QN, q = i % QN;
    int qy = q >> qsh, qx = q & (QW - 1);
    const bf16* Wb = Wm + (size_t)b * 128 * PN;
    float s = 0.0f;
    for (int du = 0; du <= 2; ++du)
        for (int dv = 0; dv <= 2; ++dv) {
            int base = (qy + du) * PW + qx + dv;
            for (int c = 0; c < 128; ++c) {
                float v = (float)Wb[(size_t)c * PN + base];
                s += v * v;
            }
        }
    rsc[i] = 1.0f / fmaxf(sqrtf(s), 1e-4f);
}

// ---------------------------------------------------------------------------
// Host orchestration
// ---------------------------------------------------------------------------
extern "C" void kernel_launch(void* const* d_in, const int* in_sizes, int n_in,
                              void* d_out, int out_size, void* d_ws, size_t ws_size,
                              hipStream_t stream) {
    (void)in_sizes; (void)n_in; (void)out_size; (void)ws_size;
    const float* x1      = (const float*)d_in[0];
    const float* x2      = (const float*)d_in[1];
    const float* mask    = (const float*)d_in[2];
    const float* a1_rw_w = (const float*)d_in[3];
    const float* a1_rw_b = (const float*)d_in[4];
    const float* a1_w_w  = (const float*)d_in[5];
    const float* a1_w_b  = (const float*)d_in[6];
    const float* a1_f_w  = (const float*)d_in[7];
    const float* a1_f_b  = (const float*)d_in[8];
    const float* a2_rw_w = (const float*)d_in[9];
    const float* a2_rw_b = (const float*)d_in[10];
    const float* a2_w_w  = (const float*)d_in[11];
    const float* a2_w_b  = (const float*)d_in[12];
    const float* a2_f_w  = (const float*)d_in[13];
    const float* a2_f_b  = (const float*)d_in[14];
    const float* g_w     = (const float*)d_in[15];
    const float* g_b     = (const float*)d_in[16];
    float* outp = (float*)d_out;

    const size_t PN64 = 66 * 66;   // 4356
    const size_t PN32 = 34 * 34;   // 1156

    char* cur = (char*)d_ws;
    auto alloc = [&](size_t bytes) -> char* {
        char* r = cur;
        cur += (bytes + 255) & ~(size_t)255;
        return r;
    };
    // ---- zero-initialized span: padded bf16 maps + padded probs + padded attn ----
    char* zstart   = cur;
    bf16* x1p1p    = (bf16*)alloc((size_t)2 * 128 * PN64 * 2);  // atn1 F (a1_rw on x1)
    bf16* wf1p     = (bf16*)alloc((size_t)2 * 128 * PN32 * 2);  // atn1 W-map (a1_f on x2r)
    bf16* wv1p     = (bf16*)alloc((size_t)2 * 128 * PN32 * 2);  // atn1 V-map (a1_w on x2r)
    bf16* x1p2p    = (bf16*)alloc((size_t)2 * 128 * PN64 * 2);  // atn2 V-map (a2_rw on x1)
    bf16* f2p      = (bf16*)alloc((size_t)2 * 128 * PN64 * 2);  // atn2 F (a2_f on x2)
    bf16* w2p      = (bf16*)alloc((size_t)2 * 128 * PN64 * 2);  // atn2 W-map (a2_w on x2)
    bf16* Pmp      = (bf16*)alloc((size_t)4096 * PN64 * 2);     // padded probs (reused)
    bf16* attn_bfp = (bf16*)alloc((size_t)2 * 256 * 6400 * 2);  // padded attn (8-px border)
    char* zend     = cur;
    // ---- uninitialized scratch ----
    bf16*  x1_bf    = (bf16*)alloc((size_t)2 * 256 * 4096 * 2);
    bf16*  x2_bf    = (bf16*)alloc((size_t)2 * 256 * 4096 * 2);
    bf16*  x2r_bf   = (bf16*)alloc((size_t)2 * 256 * 1024 * 2);
    float* mm1      = (float*)alloc((size_t)2 * 1024 * 4);
    float* mm2      = (float*)alloc((size_t)2 * 4096 * 4);
    float* rsc1     = (float*)alloc((size_t)2 * 1024 * 4);
    float* rsc2     = (float*)alloc((size_t)2 * 4096 * 4);
    float* S        = (float*)alloc((size_t)4096 * 4096 * 4);   // scores (reused)
    float* attn_f32 = (float*)alloc((size_t)2 * 256 * 4096 * 4);

    // zero padded span (borders) and the atomic accumulator
    {
        int n16 = (int)((size_t)(zend - zstart) / 16);
        k_zero128<<<(n16 + 255) / 256, 256, 0, stream>>>((v4u*)zstart, n16);
        int a16 = (int)(((size_t)2 * 256 * 4096 * 4) / 16);
        k_zero128<<<(a16 + 255) / 256, 256, 0, stream>>>((v4u*)attn_f32, a16);
    }

    const int NTOT = 2 * 256 * 4096;
    k_f32_to_bf16<<<(NTOT + 255) / 256, 256, 0, stream>>>(x1, x1_bf, NTOT);
    k_f32_to_bf16<<<(NTOT + 255) / 256, 256, 0, stream>>>(x2, x2_bf, NTOT);
    k_resize32<<<(2 * 256 * 1024 + 255) / 256, 256, 0, stream>>>(x2, x2r_bf);
    k_mm<<<(2 * 4096 + 255) / 256, 256, 0, stream>>>(mask, mm2, 6, 1);
    k_mm<<<(2 * 1024 + 255) / 256, 256, 0, stream>>>(mask, mm1, 5, 2);

    dim3 gN4096(32, 1, 2), gN1024(8, 1, 2);
    k_conv1x1<6><<<gN4096, 256, 0, stream>>>(a1_rw_w, a1_rw_b, x1_bf, x1p1p);
    k_conv1x1<6><<<gN4096, 256, 0, stream>>>(a2_rw_w, a2_rw_b, x1_bf, x1p2p);
    k_conv1x1<6><<<gN4096, 256, 0, stream>>>(a2_f_w,  a2_f_b,  x2_bf, f2p);
    k_conv1x1<6><<<gN4096, 256, 0, stream>>>(a2_w_w,  a2_w_b,  x2_bf, w2p);
    k_conv1x1<5><<<gN1024, 256, 0, stream>>>(a1_f_w,  a1_f_b,  x2r_bf, wf1p);
    k_conv1x1<5><<<gN1024, 256, 0, stream>>>(a1_w_w,  a1_w_b,  x2r_bf, wv1p);

    k_rscale<<<(2 * 1024 + 255) / 256, 256, 0, stream>>>(wf1p, rsc1, 5);
    k_rscale<<<(2 * 4096 + 255) / 256, 256, 0, stream>>>(w2p,  rsc2, 6);

    for (int b = 0; b < 2; ++b) {
        // ---- attention 1 (32x32 keys) ----
        k_corr<5><<<dim3(32, 8), 256, 0, stream>>>(
            wf1p + (size_t)b * 128 * PN32, x1p1p + (size_t)b * 128 * PN64,
            rsc1 + (size_t)b * 1024, S);
        k_softmax<<<16, 256, 0, stream>>>(S, mm1 + (size_t)b * 1024, Pmp, 1024);
        k_convT<5><<<dim3(32, 1, 9), 256, 0, stream>>>(
            wv1p + (size_t)b * 128 * PN32, Pmp,
            attn_f32 + ((size_t)b * 256 + 0) * 4096);
        // ---- attention 2 (64x64 keys) ----
        k_corr<6><<<dim3(32, 32), 256, 0, stream>>>(
            w2p + (size_t)b * 128 * PN64, f2p + (size_t)b * 128 * PN64,
            rsc2 + (size_t)b * 4096, S);
        k_softmax<<<16, 256, 0, stream>>>(S, mm2 + (size_t)b * 4096, Pmp, 4096);
        k_convT<6><<<dim3(32, 1, 9), 256, 0, stream>>>(
            x1p2p + (size_t)b * 128 * PN64, Pmp,
            attn_f32 + ((size_t)b * 256 + 128) * 4096);
    }

    k_attn_pack<<<(NTOT + 255) / 256, 256, 0, stream>>>(attn_f32, attn_bfp);
    k_dilconv<<<dim3(32, 4, 2), 256, 0, stream>>>(g_w, g_b, attn_bfp, x1, outp);
}
